// LiveSAL_80968723464405
// MI455X (gfx1250) — compile-verified, matrix operand order checked by reading, not run on hardware
//
#include <hip/hip_runtime.h>
#include <hip/hip_bf16.h>
#include <cstdint>
#include <cstddef>

// Problem constants (from reference)
#define S_  5
#define B_  2
#define C_  128
#define H_  24
#define W_  24
#define N_  576            // H*W
#define NC  (N_*C_)        // 73728
#define NN  ((size_t)N_*N_)// 331776
#define SB  (S_*B_)        // 10
#define SSB (S_*S_*B_)     // 50
#define K3  (C_*9)         // 1152
#define C2  (2*C_)         // 256
#define K5  (C2*25)        // 6400
#define NITER 3

typedef __attribute__((ext_vector_type(16))) __bf16 bf16x16;
typedef __attribute__((ext_vector_type(8)))  __bf16 bf16x8;
typedef __attribute__((ext_vector_type(8)))  float  f32x8;

__device__ __forceinline__ unsigned short f2bf(float f) {
  union { float f; unsigned u; } x; x.f = f;
  unsigned r = x.u + 0x7FFFu + ((x.u >> 16) & 1u);
  return (unsigned short)(r >> 16);
}

// ---------------------------------------------------------------------------
// Generic NT GEMM on WMMA bf16 -> f32.
//   D[m][n] = act( sum_k A[m][k]*B[n][k] + bias[m] )
// A: (M x K) row-major bf16, B: (N x K) row-major bf16 (i.e. B^T k-contiguous)
// Block = 128 threads = 4 waves; block tile 64(M) x 64(N); wave tile 16 x 64.
// The 64x32 B slab (shared by all 4 waves) is double-buffered in LDS, filled
// with GLOBAL_LOAD_ASYNC_TO_LDS_B128 (ASYNCcnt), consumed via ds_load_b128.
// transposed=1 stores D[n*M + m] (gives (N,M) layout directly).
// outF (f32) and/or outB (bf16) may be null. act: 0=none 1=sigmoid 2=tanh.
// ---------------------------------------------------------------------------
__global__ __launch_bounds__(128)
void k_gemm_nt(const unsigned short* __restrict__ A, size_t strA,
               const unsigned short* __restrict__ B, size_t strB,
               float* __restrict__ outF, unsigned short* __restrict__ outB,
               size_t strC, const float* __restrict__ bias,
               int M, int N, int K, int transposed, int act)
{
  __shared__ unsigned short shB[2][64 * 32];   // 2 x 4KB B slabs

  const int batch = blockIdx.y;
  const __bf16* Ab = reinterpret_cast<const __bf16*>(A) + (size_t)batch * strA;
  const __bf16* Bb = reinterpret_cast<const __bf16*>(B) + (size_t)batch * strB;
  const int tilesN = N >> 6;
  const int tm = blockIdx.x / tilesN;
  const int tn = blockIdx.x % tilesN;
  const int tid  = threadIdx.x;
  const int wave = tid >> 5;
  const int lane = tid & 31;
  const int m0 = (tm << 6) + (wave << 4);
  const int n0 = tn << 6;
  const int lrow = lane & 15;
  const int kh8  = (lane >> 4) << 3;   // 0 or 8 (per-half K offset)

  const __bf16* pa = Ab + (size_t)(m0 + lrow) * K + kh8;

  // Stage the 64(N-rows) x 32(K) B slab for k0 into LDS buffer `buf`.
  // 64 rows x 64B = 4KB = 128 threads x 2 x b128 async transfers.
  auto stage = [&](int buf, int k0) {
#pragma unroll
    for (int t = 0; t < 2; ++t) {
      const int id    = tid + (t << 7);
      const int row   = id >> 2;        // 0..63
      const int chunk = id & 3;         // 0..3 (16B each)
      const __bf16* g = Bb + (size_t)(n0 + row) * K + k0 + (chunk << 3);
      const unsigned lds =
          (unsigned)(size_t)(const void*)&shB[buf][row * 32 + (chunk << 3)];
      asm volatile("global_load_async_to_lds_b128 %0, %1, off"
                   :: "v"(lds), "v"((unsigned long long)(size_t)g)
                   : "memory");
    }
  };

  f32x8 acc0 = {}, acc1 = {}, acc2 = {}, acc3 = {};
  const int nsteps = K >> 5;

  stage(0, 0);
  for (int s = 0; s < nsteps; ++s) {
    const int k0 = s << 5;
    if (s + 1 < nsteps) {
      stage((s + 1) & 1, k0 + 32);
      asm volatile("s_wait_asynccnt 0x2" ::: "memory");  // stage s complete
    } else {
      asm volatile("s_wait_asynccnt 0x0" ::: "memory");
    }
    __syncthreads();

    const unsigned short* sb = shB[s & 1];
    union U { bf16x16 v; bf16x8 h[2]; } a, b0, b1, b2, b3;
    a.h[0] = *reinterpret_cast<const bf16x8*>(pa + k0);
    a.h[1] = *reinterpret_cast<const bf16x8*>(pa + k0 + 16);
    __builtin_prefetch((const void*)(pa + k0 + 256), 0, 1);
    b0.h[0] = *reinterpret_cast<const bf16x8*>(&sb[( 0 + lrow) * 32 + kh8]);
    b0.h[1] = *reinterpret_cast<const bf16x8*>(&sb[( 0 + lrow) * 32 + 16 + kh8]);
    b1.h[0] = *reinterpret_cast<const bf16x8*>(&sb[(16 + lrow) * 32 + kh8]);
    b1.h[1] = *reinterpret_cast<const bf16x8*>(&sb[(16 + lrow) * 32 + 16 + kh8]);
    b2.h[0] = *reinterpret_cast<const bf16x8*>(&sb[(32 + lrow) * 32 + kh8]);
    b2.h[1] = *reinterpret_cast<const bf16x8*>(&sb[(32 + lrow) * 32 + 16 + kh8]);
    b3.h[0] = *reinterpret_cast<const bf16x8*>(&sb[(48 + lrow) * 32 + kh8]);
    b3.h[1] = *reinterpret_cast<const bf16x8*>(&sb[(48 + lrow) * 32 + 16 + kh8]);

    acc0 = __builtin_amdgcn_wmma_f32_16x16x32_bf16(false, a.v, false, b0.v, (short)0, acc0, false, false);
    acc1 = __builtin_amdgcn_wmma_f32_16x16x32_bf16(false, a.v, false, b1.v, (short)0, acc1, false, false);
    acc2 = __builtin_amdgcn_wmma_f32_16x16x32_bf16(false, a.v, false, b2.v, (short)0, acc2, false, false);
    acc3 = __builtin_amdgcn_wmma_f32_16x16x32_bf16(false, a.v, false, b3.v, (short)0, acc3, false, false);
    __syncthreads();   // protect slab before it is re-staged (stage s+2)
  }

  const int mbase = m0 + ((lane >> 4) << 3);
  for (int r = 0; r < 8; ++r) {
    const int m = mbase + r;
    const float bv = bias ? bias[m] : 0.0f;
    float vals[4] = { acc0[r], acc1[r], acc2[r], acc3[r] };
    for (int j = 0; j < 4; ++j) {
      float v = vals[j] + bv;
      if (act == 1)      v = 1.0f / (1.0f + __expf(-v));
      else if (act == 2) v = tanhf(v);
      const int n = n0 + (j << 4) + lrow;
      const size_t idx = transposed ? ((size_t)n * M + m) : ((size_t)m * N + n);
      const size_t off = (size_t)batch * strC + idx;
      if (outF) outF[off] = v;
      if (outB) outB[off] = f2bf(v);
    }
  }
}

// im2col for (batch, Cin, 24x24) fp32 source -> (batch, N, K) bf16 patches
__global__ void k_im2col(const float* __restrict__ src, size_t strS,
                         unsigned short* __restrict__ dst, size_t strD,
                         int KH, int KW, int pad, int Kk)
{
  const int batch = blockIdx.y;
  const int idx = blockIdx.x * blockDim.x + threadIdx.x;
  if (idx >= N_ * Kk) return;
  const int n = idx / Kk;
  const int k = idx - n * Kk;
  const int khw = KH * KW;
  const int c = k / khw;
  const int r = k - c * khw;
  const int kh = r / KW, kw = r - kh * KW;
  const int hh = n / W_ + kh - pad;
  const int ww = n % W_ + kw - pad;
  float v = 0.0f;
  if (hh >= 0 && hh < H_ && ww >= 0 && ww < W_)
    v = src[(size_t)batch * strS + (size_t)c * N_ + hh * W_ + ww];
  dst[(size_t)batch * strD + (size_t)n * Kk + k] = f2bf(v);
}

// Row softmax (ncols), fp32 in -> normalized bf16 out
__global__ __launch_bounds__(256)
void k_softmax_bf(const float* __restrict__ in, unsigned short* __restrict__ out, int ncols)
{
  __shared__ float red[256];
  const size_t row = blockIdx.x;
  const float* p = in + row * (size_t)ncols;
  float lmax = -3.0e38f;
  for (int c = threadIdx.x; c < ncols; c += 256) lmax = fmaxf(lmax, p[c]);
  red[threadIdx.x] = lmax; __syncthreads();
  for (int s = 128; s > 0; s >>= 1) {
    if ((int)threadIdx.x < s) red[threadIdx.x] = fmaxf(red[threadIdx.x], red[threadIdx.x + s]);
    __syncthreads();
  }
  const float mx = red[0]; __syncthreads();
  float lsum = 0.0f;
  for (int c = threadIdx.x; c < ncols; c += 256) lsum += __expf(p[c] - mx);
  red[threadIdx.x] = lsum; __syncthreads();
  for (int s = 128; s > 0; s >>= 1) {
    if ((int)threadIdx.x < s) red[threadIdx.x] += red[threadIdx.x + s];
    __syncthreads();
  }
  const float inv = 1.0f / red[0];
  unsigned short* q = out + row * (size_t)ncols;
  for (int c = threadIdx.x; c < ncols; c += 256) q[c] = f2bf(__expf(p[c] - mx) * inv);
}

__global__ void k_cast_bf(const float* __restrict__ s, unsigned short* __restrict__ d, int n)
{ int i = blockIdx.x * blockDim.x + threadIdx.x; if (i < n) d[i] = f2bf(s[i]); }

__global__ void k_copy(const float* __restrict__ s, float* __restrict__ d, int n)
{ int i = blockIdx.x * blockDim.x + threadIdx.x; if (i < n) d[i] = s[i]; }

// h (b, C, N) fp32 -> (b, N, C) bf16
__global__ void k_transpose_cast(const float* __restrict__ s, unsigned short* __restrict__ d)
{
  int idx = blockIdx.x * blockDim.x + threadIdx.x;
  if (idx >= SB * NC) return;
  const int n = idx % N_;
  const int c = (idx / N_) % C_;
  const int b = idx / (C_ * N_);
  d[(size_t)b * NC + (size_t)n * C_ + c] = f2bf(s[idx]);
}

// colsum[b][k] = sum over valid spatial of shifted window, from mjiT (b, N, C)
__global__ void k_gate_colsum(const float* __restrict__ mjiT, float* __restrict__ colsum)
{
  int idx = blockIdx.x * blockDim.x + threadIdx.x;
  if (idx >= SSB * K3) return;
  const int k = idx % K3;
  const int b = idx / K3;
  const int c = k / 9;
  const int r = k % 9;
  const int kh = r / 3, kw = r % 3;
  const float* base = mjiT + (size_t)b * NC;
  float s = 0.0f;
  for (int hh = 0; hh < H_; ++hh) {
    const int ih = hh + kh - 1;
    if (ih < 0 || ih >= H_) continue;
    for (int ww = 0; ww < W_; ++ww) {
      const int iw = ww + kw - 1;
      if (iw < 0 || iw >= W_) continue;
      s += base[(size_t)(ih * W_ + iw) * C_ + c];
    }
  }
  colsum[idx] = s;
}

// g[b][c] = sigmoid( mean-of-conv ) = sigmoid( dot(Wg[c], colsum[b]) / N + bias )
__global__ void k_gate(const float* __restrict__ colsum, const float* __restrict__ Wg_w,
                       const float* __restrict__ Wg_b, float* __restrict__ g)
{
  int idx = blockIdx.x * blockDim.x + threadIdx.x;
  if (idx >= SSB * C_) return;
  const int c = idx % C_;
  const int b = idx / C_;
  const float* wrow = Wg_w + (size_t)c * K3;
  const float* cs = colsum + (size_t)b * K3;
  float s = 0.0f;
  for (int k = 0; k < K3; ++k) s += wrow[k] * cs[k];
  g[idx] = 1.0f / (1.0f + __expf(-(s * (1.0f / (float)N_) + Wg_b[c])));
}

// xi = intra * (alpha*ctx + h) + interw * (bn_scale * sum_{j!=i} g*mji + (S-1)*bn_beta)
__global__ void k_inter_msg(const float* __restrict__ g, const float* __restrict__ mjiT,
                            const float* __restrict__ ctxT, const float* __restrict__ h,
                            const float* __restrict__ alpha, const float* __restrict__ bn_gamma,
                            const float* __restrict__ bn_beta, const float* __restrict__ intra_w,
                            const float* __restrict__ inter_w, float* __restrict__ xi)
{
  int idx = blockIdx.x * blockDim.x + threadIdx.x;
  if (idx >= SB * C_ * N_) return;
  const int n = idx % N_;
  const int c = (idx / N_) % C_;
  const int sb = idx / (N_ * C_);   // i*B_ + b
  const int i = sb / B_, b = sb % B_;
  const float bns = bn_gamma[c] * rsqrtf(1.0f + 1e-5f);
  float acc = 0.0f;
  for (int j = 0; j < S_; ++j) {
    if (j == i) continue;
    const int ijb = (i * S_ + j) * B_ + b;
    acc += g[ijb * C_ + c] * mjiT[(size_t)ijb * NC + (size_t)n * C_ + c];
  }
  const float inter = bns * acc + (float)(S_ - 1) * bn_beta[c];
  const float eii = alpha[0] * ctxT[(size_t)sb * NC + (size_t)n * C_ + c] + h[idx];
  xi[idx] = intra_w[0] * eii + inter_w[0] * inter;
}

// cat = concat([xi, h]) or concat([xi, r*h]) along channels (b, 2C, N)
__global__ void k_concat(const float* __restrict__ xi, const float* __restrict__ h,
                         const float* __restrict__ zr, float* __restrict__ cat, int useR)
{
  int idx = blockIdx.x * blockDim.x + threadIdx.x;
  if (idx >= SB * C2 * N_) return;
  const int n = idx % N_;
  const int c = (idx / N_) % C2;
  const int b = idx / (N_ * C2);
  float v;
  if (c < C_) {
    v = xi[(size_t)b * C_ * N_ + (size_t)c * N_ + n];
  } else {
    v = h[(size_t)b * C_ * N_ + (size_t)(c - C_) * N_ + n];
    if (useR) v *= zr[(size_t)b * C2 * N_ + (size_t)c * N_ + n];  // r channel = C_+(c-C_)
  }
  cat[idx] = v;
}

// h = (1-z)*h + z*h_hat + h
__global__ void k_gru_update(const float* __restrict__ zr, const float* __restrict__ hhat,
                             float* __restrict__ h)
{
  int idx = blockIdx.x * blockDim.x + threadIdx.x;
  if (idx >= SB * C_ * N_) return;
  const int n = idx % N_;
  const int c = (idx / N_) % C_;
  const int b = idx / (N_ * C_);
  const float z = zr[(size_t)b * C2 * N_ + (size_t)c * N_ + n];
  const float hv = h[idx];
  h[idx] = (1.0f - z) * hv + z * hhat[idx] + hv;
}

// ---------------------------------------------------------------------------
extern "C" void kernel_launch(void* const* d_in, const int* in_sizes, int n_in,
                              void* d_out, int out_size, void* d_ws, size_t ws_size,
                              hipStream_t stream)
{
  (void)in_sizes; (void)n_in; (void)out_size; (void)ws_size;
  const float* x        = (const float*)d_in[0];
  const float* Wf_w     = (const float*)d_in[1];
  const float* Wf_b     = (const float*)d_in[2];
  const float* Wh_w     = (const float*)d_in[3];
  const float* Wh_b     = (const float*)d_in[4];
  const float* Wl_w     = (const float*)d_in[5];
  const float* Wl_b     = (const float*)d_in[6];
  const float* alpha    = (const float*)d_in[7];
  const float* Wc       = (const float*)d_in[8];
  const float* Wg_w     = (const float*)d_in[9];
  const float* Wg_b     = (const float*)d_in[10];
  const float* bn_gamma = (const float*)d_in[11];
  const float* bn_beta  = (const float*)d_in[12];
  const float* gru_zr_w = (const float*)d_in[13];
  const float* gru_zr_b = (const float*)d_in[14];
  const float* gru_h_w  = (const float*)d_in[15];
  const float* gru_h_b  = (const float*)d_in[16];
  const float* intra_w  = (const float*)d_in[17];
  const float* inter_w  = (const float*)d_in[18];
  float* out = (float*)d_out;

  // workspace carve-up (256B aligned)
  char* w = (char*)d_ws;
  size_t off = 0;
  auto alloc = [&](size_t bytes) -> void* {
    off = (off + 255) & ~(size_t)255;
    void* p = w + off;
    off += bytes;
    return p;
  };
  unsigned short* wfB  = (unsigned short*)alloc((size_t)C_ * K3 * 2);
  unsigned short* whB  = (unsigned short*)alloc((size_t)C_ * K3 * 2);
  unsigned short* wlB  = (unsigned short*)alloc((size_t)C_ * K3 * 2);
  unsigned short* wcB  = (unsigned short*)alloc((size_t)C_ * C_ * 2);
  unsigned short* wzrB = (unsigned short*)alloc((size_t)C2 * K5 * 2);
  unsigned short* wghB = (unsigned short*)alloc((size_t)C_ * K5 * 2);
  float*          hbuf = (float*)alloc((size_t)SB * NC * 4);
  unsigned short* hbf  = (unsigned short*)alloc((size_t)SB * NC * 2);
  unsigned short* hbfT = (unsigned short*)alloc((size_t)SB * NC * 2);
  unsigned short* col3 = (unsigned short*)alloc((size_t)SB * N_ * K3 * 2);
  unsigned short* ftT  = (unsigned short*)alloc((size_t)SB * NC * 2);
  unsigned short* htT  = (unsigned short*)alloc((size_t)SB * NC * 2);
  unsigned short* ltB  = (unsigned short*)alloc((size_t)SB * NC * 2);
  float*          logA = (float*)alloc((size_t)SB * NN * 4);
  unsigned short* attB = (unsigned short*)alloc((size_t)SB * NN * 2);
  float*          ctxT = (float*)alloc((size_t)SB * NC * 4);
  unsigned short* ytT  = (unsigned short*)alloc((size_t)SB * NC * 2);
  float*          logE = (float*)alloc((size_t)SSB * NN * 4);
  unsigned short* eijB = (unsigned short*)alloc((size_t)SSB * NN * 2);
  float*          mjiT = (float*)alloc((size_t)SSB * NC * 4);
  float*          csum = (float*)alloc((size_t)SSB * K3 * 4);
  float*          gbuf = (float*)alloc((size_t)SSB * C_ * 4);
  float*          xi   = (float*)alloc((size_t)SB * NC * 4);
  float*          cat  = (float*)alloc((size_t)SB * C2 * N_ * 4);
  unsigned short* col5 = (unsigned short*)alloc((size_t)SB * N_ * K5 * 2);
  float*          zr   = (float*)alloc((size_t)SB * C2 * N_ * 4);
  float*          hhat = (float*)alloc((size_t)SB * C_ * N_ * 4);

  auto cast = [&](const float* s, unsigned short* d, int n) {
    k_cast_bf<<<dim3((n + 255) / 256), 256, 0, stream>>>(s, d, n);
  };
  auto gemm = [&](const void* A, size_t sA, const void* Bm, size_t sBm,
                  void* oF, void* oB, size_t sC, const float* bias,
                  int M, int N, int K, int transposed, int act, int batch) {
    dim3 grid((unsigned)((M >> 6) * (N >> 6)), (unsigned)batch);
    k_gemm_nt<<<grid, 128, 0, stream>>>((const unsigned short*)A, sA,
                                        (const unsigned short*)Bm, sBm,
                                        (float*)oF, (unsigned short*)oB, sC, bias,
                                        M, N, K, transposed, act);
  };

  // one-time weight conversion (re-done each call: deterministic, cheap)
  cast(Wf_w, wfB, C_ * K3);
  cast(Wh_w, whB, C_ * K3);
  cast(Wl_w, wlB, C_ * K3);
  cast(Wc,   wcB, C_ * C_);
  cast(gru_zr_w, wzrB, C2 * K5);
  cast(gru_h_w,  wghB, C_ * K5);

  const int HN = SB * NC;  // 737280 elements of h
  k_copy<<<dim3((HN + 255) / 256), 256, 0, stream>>>(x, hbuf, HN);

  for (int it = 0; it < NITER; ++it) {
    cast(hbuf, hbf, HN);
    k_transpose_cast<<<dim3((HN + 255) / 256), 256, 0, stream>>>(hbuf, hbfT);

    // --- 3x3 convs as implicit GEMM: ft/ht -> (N,C) bf16, lt -> (C,N) bf16
    {
      dim3 grid((N_ * K3 + 255) / 256, SB);
      k_im2col<<<grid, 256, 0, stream>>>(hbuf, (size_t)C_ * N_, col3, (size_t)N_ * K3, 3, 3, 1, K3);
    }
    gemm(wfB, 0, col3, (size_t)N_ * K3, nullptr, ftT, NC, Wf_b, C_, N_, K3, 1, 0, SB);
    gemm(whB, 0, col3, (size_t)N_ * K3, nullptr, htT, NC, Wh_b, C_, N_, K3, 1, 0, SB);
    gemm(wlB, 0, col3, (size_t)N_ * K3, nullptr, ltB, NC, Wl_b, C_, N_, K3, 0, 0, SB);

    // --- spatial attention: logits[n][m] = ft.h  -> softmax -> ctxT
    gemm(ftT, NC, htT, NC, logA, nullptr, NN, nullptr, N_, N_, C_, 0, 0, SB);
    k_softmax_bf<<<dim3(SB * N_), 256, 0, stream>>>(logA, attB, N_);
    gemm(attB, NN, ltB, NC, ctxT, nullptr, NC, nullptr, N_, C_, N_, 0, 0, SB);

    // --- yt = Wc @ h  -> (N,C) bf16
    gemm(wcB, 0, hbfT, NC, nullptr, ytT, NC, nullptr, C_, N_, C_, 1, 0, SB);

    // --- cross-image attention eij[i,j] and mji[i,j]
    for (int i = 0; i < S_; ++i)
      for (int j = 0; j < S_; ++j) {
        gemm(hbfT + (size_t)i * B_ * NC, NC,
             ytT  + (size_t)j * B_ * NC, NC,
             logE + (size_t)((i * S_ + j) * B_) * NN, nullptr, NN, nullptr,
             N_, N_, C_, 0, 0, B_);
      }
    k_softmax_bf<<<dim3(SSB * N_), 256, 0, stream>>>(logE, eijB, N_);
    for (int i = 0; i < S_; ++i)
      for (int j = 0; j < S_; ++j) {
        gemm(eijB + (size_t)((i * S_ + j) * B_) * NN, NN,
             hbf  + (size_t)j * B_ * NC, NC,
             mjiT + (size_t)((i * S_ + j) * B_) * NC, nullptr, NC, nullptr,
             N_, C_, N_, 0, 0, B_);
      }

    // --- gate (conv+mean collapsed to colsum + matvec), then fuse messages
    k_gate_colsum<<<dim3((SSB * K3 + 255) / 256), 256, 0, stream>>>(mjiT, csum);
    k_gate<<<dim3((SSB * C_ + 255) / 256), 256, 0, stream>>>(csum, Wg_w, Wg_b, gbuf);
    k_inter_msg<<<dim3((HN + 255) / 256), 256, 0, stream>>>(gbuf, mjiT, ctxT, hbuf,
                                                            alpha, bn_gamma, bn_beta,
                                                            intra_w, inter_w, xi);

    // --- GRU: zr = sigmoid(conv5(concat[xi,h]))
    const int CN2 = SB * C2 * N_;
    k_concat<<<dim3((CN2 + 255) / 256), 256, 0, stream>>>(xi, hbuf, nullptr, cat, 0);
    {
      dim3 grid((N_ * K5 + 255) / 256, SB);
      k_im2col<<<grid, 256, 0, stream>>>(cat, (size_t)C2 * N_, col5, (size_t)N_ * K5, 5, 5, 2, K5);
    }
    gemm(wzrB, 0, col5, (size_t)N_ * K5, zr, nullptr, (size_t)C2 * N_, gru_zr_b,
         C2, N_, K5, 0, 1, SB);

    // --- h_hat = tanh(conv5(concat[xi, r*h]))
    k_concat<<<dim3((CN2 + 255) / 256), 256, 0, stream>>>(xi, hbuf, zr, cat, 1);
    {
      dim3 grid((N_ * K5 + 255) / 256, SB);
      k_im2col<<<grid, 256, 0, stream>>>(cat, (size_t)C2 * N_, col5, (size_t)N_ * K5, 5, 5, 2, K5);
    }
    gemm(wghB, 0, col5, (size_t)N_ * K5, hhat, nullptr, (size_t)C_ * N_, gru_h_b,
         C_, N_, K5, 0, 2, SB);

    // --- h = (1-z)*h + z*h_hat + h
    k_gru_update<<<dim3((HN + 255) / 256), 256, 0, stream>>>(zr, hhat, hbuf);
  }

  k_copy<<<dim3((HN + 255) / 256), 256, 0, stream>>>(hbuf, out, HN);
}
